// FixSetLinearAttention_63513976373589
// MI455X (gfx1250) — matrix operand order repaired and checked by reading, not compiled, not run on hardware
//
#include <hip/hip_runtime.h>
#include <stdint.h>

typedef __attribute__((ext_vector_type(16))) __bf16 v16bf;
typedef __attribute__((ext_vector_type(8)))  float  v8f;
typedef uint32_t u32;
typedef uint16_t u16;

#define NHEAD 12
#define HS    64
#define TSEQ  2048
#define BB    8
#define CDIM  768
#define SSEG  255
#define SPAD  256
#define INFTY __builtin_inff()

// ---------- bf16 helpers (round-to-nearest-even) ----------
__device__ __forceinline__ u16 bf16rn(float f) {
  u32 x = __builtin_bit_cast(u32, f);
  x += 0x7FFFu + ((x >> 16) & 1u);
  return (u16)(x >> 16);
}
__device__ __forceinline__ u32 pk2(float lo, float hi) {
  return (u32)bf16rn(lo) | ((u32)bf16rn(hi) << 16);
}
__device__ __forceinline__ float bf2f(u16 h) {
  return __builtin_bit_cast(float, (u32)h << 16);
}

struct F32B { uint4 lo, hi; };

// ---------- WMMA fragment loaders (wave32, 16x16x32 bf16) ----------
// A-matrix 16xK tile, row-major bf16 in LDS. Contiguous k order used here:
// lanes 0-15: K = c*32 + 0..15; lanes 16-31: +16 (consecutive).
__device__ __forceinline__ v16bf fragA_lds(const u16* tile, int ldk, int kc) {
  int lane = threadIdx.x & 31;
  const u16* p = tile + (lane & 15) * ldk + kc * 32 + ((lane >> 4) << 4);
  F32B f; f.lo = *(const uint4*)p; f.hi = *(const uint4*)(p + 8);
  return __builtin_bit_cast(v16bf, f);
}
// B-matrix (KxN) supplied as B^T row-major [N=16][K]. Same contiguous k order as fragA:
// lane = column, lanes 0-15 hold K = c*32+0..15, lanes 16-31 hold +16 (consecutive).
__device__ __forceinline__ v16bf fragB_lds(const u16* tileT, int ldk, int kc) {
  int lane = threadIdx.x & 31;
  const u16* p = tileT + (lane & 15) * ldk + kc * 32 + ((lane >> 4) << 4);
  F32B f; f.lo = *(const uint4*)p; f.hi = *(const uint4*)(p + 8);
  return __builtin_bit_cast(v16bf, f);
}
__device__ __forceinline__ v16bf fragB_gbf(const u16* g, int ldk, int kbase) {
  int lane = threadIdx.x & 31;
  const u16* p = g + (long)(lane & 15) * ldk + kbase + ((lane >> 4) << 4);
  F32B f; f.lo = *(const uint4*)p; f.hi = *(const uint4*)(p + 8);
  return __builtin_bit_cast(v16bf, f);
}
__device__ __forceinline__ v16bf fragB_gf32(const float* g, int ldk, int kbase) {
  int lane = threadIdx.x & 31;
  const float* p = g + (long)(lane & 15) * ldk + kbase + ((lane >> 4) << 4);
  float4 a = *(const float4*)p,      b = *(const float4*)(p + 4);
  float4 c = *(const float4*)(p + 8), d = *(const float4*)(p + 12);
  F32B f;
  f.lo = make_uint4(pk2(a.x,a.y), pk2(a.z,a.w), pk2(b.x,b.y), pk2(b.z,b.w));
  f.hi = make_uint4(pk2(c.x,c.y), pk2(c.z,c.w), pk2(d.x,d.y), pk2(d.z,d.w));
  return __builtin_bit_cast(v16bf, f);
}
__device__ __forceinline__ v8f wmma_bf16(v16bf a, v16bf b, v8f c) {
  return __builtin_amdgcn_wmma_f32_16x16x32_bf16(false, a, false, b, (short)0, c,
                                                 false, false);
}

// ---------- K0: f32 -> bf16 weight conversion ----------
__global__ void k_f32_to_bf16(const float* __restrict__ in, u16* __restrict__ out,
                              int n) {
  int i = blockIdx.x * blockDim.x + threadIdx.x;
  if (i < n) out[i] = bf16rn(in[i]);
}

// ---------- K1/K5: C[M,N] = A[M,K] @ B[N,K]^T  (bf16 WMMA, f32 out) ----------
__global__ __launch_bounds__(256)
void k_gemm_bt(const float* __restrict__ A, const float* __restrict__ B,
               float* __restrict__ C, int K, int lda, int ldb, int ldc) {
  __shared__ u16 At[16 * 64];
  __shared__ u16 Bt[128 * 64];
  int m0 = blockIdx.x * 16;
  int n0 = blockIdx.y * 128;
  int tid = threadIdx.x;
  int w = tid >> 5;
  v8f acc = {};
  for (int k0 = 0; k0 < K; k0 += 64) {
    if (k0 + 64 < K)  // warm next chunk of B into caches
      __builtin_prefetch(B + (long)(n0 + (tid >> 1)) * ldb + k0 + 64, 0, 1);
#pragma unroll
    for (int i = 0; i < 2; ++i) {            // stage A (16x64 f32 -> bf16)
      int idx = tid + i * 256;
      int r = idx >> 5, c2 = (idx & 31) * 2;
      float2 v = *(const float2*)(A + (long)(m0 + r) * lda + k0 + c2);
      *(u32*)(At + r * 64 + c2) = pk2(v.x, v.y);
    }
#pragma unroll
    for (int i = 0; i < 16; ++i) {           // stage B^T (128x64 f32 -> bf16)
      int idx = tid + i * 256;
      int r = idx >> 5, c2 = (idx & 31) * 2;
      float2 v = *(const float2*)(B + (long)(n0 + r) * ldb + k0 + c2);
      *(u32*)(Bt + r * 64 + c2) = pk2(v.x, v.y);
    }
    __syncthreads();
    v16bf a0 = fragA_lds(At, 64, 0);
    v16bf a1 = fragA_lds(At, 64, 1);
    v16bf b0 = fragB_lds(Bt + w * 16 * 64, 64, 0);
    v16bf b1 = fragB_lds(Bt + w * 16 * 64, 64, 1);
    acc = wmma_bf16(a0, b0, acc);
    acc = wmma_bf16(a1, b1, acc);
    __syncthreads();
  }
  int lane = tid & 31, col = lane & 15, rbase = (lane >> 4) << 3;
#pragma unroll
  for (int r = 0; r < 8; ++r)
    C[(long)(m0 + rbase + r) * ldc + n0 + w * 16 + col] = acc[r];
}

// ---------- K2: v[t,e] = sum_ij k_i v0_j Wv[e, i*64+j] + bv[e] ----------
#define KCV 1024
__global__ __launch_bounds__(128)
void k_vproj(const float* __restrict__ qkv, const u16* __restrict__ Wvb,
             const float* __restrict__ bv, float* __restrict__ vout) {
  __shared__ float kt[16][64];
  __shared__ float v0t[16][64];
  __shared__ u16 P[16 * KCV];
  int blk = blockIdx.x;
  int tt = blk & (TSEQ / 16 - 1);
  int bh = blk >> 7;                 // T/16 == 128
  int b = bh / NHEAD, h = bh % NHEAD;
  int t0 = tt * 16;
  int tid = threadIdx.x;
  for (int idx = tid; idx < 16 * 64; idx += 128) {
    int r = idx >> 6, d = idx & 63;
    long row = (long)(b * TSEQ + t0 + r) * (3 * CDIM);
    kt[r][d]  = qkv[row + CDIM + h * HS + d];
    v0t[r][d] = qkv[row + 2 * CDIM + h * HS + d];
  }
  __syncthreads();
  int w = tid >> 5;
  v8f acc = {};
  for (int cc = 0; cc < HS * HS; cc += KCV) {
    for (int idx = tid; idx < 16 * KCV / 2; idx += 128) {   // build P chunk
      int r = idx >> 9;
      int c2 = (idx & 511) * 2;
      int cg = cc + c2;
      int i = cg >> 6, j = cg & 63;
      float kk = kt[r][i];
      *(u32*)(P + r * KCV + c2) = pk2(kk * v0t[r][j], kk * v0t[r][j + 1]);
    }
    __syncthreads();
    const u16* brow = Wvb + (long)(w * 16) * (HS * HS) + cc;
#pragma unroll 4
    for (int s = 0; s < KCV / 32; ++s) {
      v16bf a = fragA_lds(P, KCV, s);
      v16bf bfr = fragB_gbf(brow, HS * HS, s * 32);
      acc = wmma_bf16(a, bfr, acc);
    }
    __syncthreads();
  }
  int lane = tid & 31, col = lane & 15, rbase = (lane >> 4) << 3;
  float bvv = bv[w * 16 + col];
#pragma unroll
  for (int r = 0; r < 8; ++r)
    vout[((long)bh * TSEQ + t0 + rbase + r) * HS + w * 16 + col] = acc[r] + bvv;
}

// ---------- K3: dyadic segment sums (255 segs + zero pad) ----------
__global__ __launch_bounds__(256)
void k_segsum(const float* __restrict__ qkv, const float* __restrict__ v,
              float* __restrict__ Kseg, float* __restrict__ VsegT) {
  __shared__ float S[SPAD][64];
  int bh = blockIdx.x;
  int b = bh / NHEAD, h = bh % NHEAD;
  int tid = threadIdx.x;
  int d = tid & 63, g = tid >> 6;
  // ---- K ----
  {
    const float* base = qkv + (long)b * TSEQ * (3 * CDIM) + CDIM + h * HS + d;
    for (int s = g; s < 128; s += 4) {
      float sum = 0.f;
#pragma unroll
      for (int r = 0; r < 16; ++r) sum += base[(long)(s * 16 + r) * (3 * CDIM)];
      S[s][d] = sum;
    }
    __syncthreads();
    int prev = 0, cur = 128, n = 64;
    while (n >= 1) {
      for (int idx = tid; idx < n * 64; idx += 256) {
        int s = idx >> 6, dd = idx & 63;
        S[cur + s][dd] = S[prev + 2 * s][dd] + S[prev + 2 * s + 1][dd];
      }
      __syncthreads();
      prev = cur; cur += n; n >>= 1;
    }
    float* out = Kseg + (long)bh * SPAD * 64;
    for (int idx = tid; idx < SPAD * 64; idx += 256) {
      int s = idx >> 6, dd = idx & 63;
      out[idx] = (s < SSEG) ? S[s][dd] : 0.f;
    }
    __syncthreads();
  }
  // ---- V (write transposed: [e][s]) ----
  {
    const float* base = v + (long)bh * TSEQ * HS + d;
    for (int s = g; s < 128; s += 4) {
      float sum = 0.f;
#pragma unroll
      for (int r = 0; r < 16; ++r) sum += base[(s * 16 + r) * HS];
      S[s][d] = sum;
    }
    __syncthreads();
    int prev = 0, cur = 128, n = 64;
    while (n >= 1) {
      for (int idx = tid; idx < n * 64; idx += 256) {
        int s = idx >> 6, dd = idx & 63;
        S[cur + s][dd] = S[prev + 2 * s][dd] + S[prev + 2 * s + 1][dd];
      }
      __syncthreads();
      prev = cur; cur += n; n >>= 1;
    }
    float* out = VsegT + (long)bh * 64 * SPAD;
    for (int idx = tid; idx < 64 * SPAD; idx += 256) {
      int dd = idx >> 8, s = idx & 255;
      out[idx] = (s < SSEG) ? S[s][dd] : 0.f;
    }
  }
}

// ---------- K4: masked segment attention + in-tile tail, fused softmax ----------
__global__ __launch_bounds__(32)
void k_attn(const float* __restrict__ qkv, const float* __restrict__ v,
            const float* __restrict__ Kseg, const float* __restrict__ VsegT,
            float* __restrict__ aout) {
  __shared__ u16 qb[16 * 64];
  __shared__ float kc[16][64];
  __shared__ float vc[16][64];
  __shared__ u16 attw[16 * SPAD];
  __shared__ float tailL[16];
  __shared__ float tailW[16];
  int bh = blockIdx.x;
  int b = bh / NHEAD, h = bh % NHEAD;
  int t0 = blockIdx.y * 16;
  int lane = threadIdx.x;
  const float scale = 0.125f;  // 1/sqrt(64)
  for (int idx = lane; idx < 16 * 64; idx += 32) {
    int r = idx >> 6, d = idx & 63;
    long row = (long)(b * TSEQ + t0 + r) * (3 * CDIM);
    qb[idx]  = bf16rn(qkv[row + h * HS + d]);
    kc[r][d] = qkv[row + CDIM + h * HS + d];
    vc[r][d] = v[((long)bh * TSEQ + t0 + r) * HS + d];
  }
  __syncthreads();
  // in-tile prefix sums: K_tail / V_tail (l_tail = tile start since tiles = Lmin)
#pragma unroll
  for (int cpl = 0; cpl < 2; ++cpl) {
    int d = lane + cpl * 32;
    float rk = 0.f, rv = 0.f;
#pragma unroll
    for (int t = 0; t < 16; ++t) {
      rk += kc[t][d]; kc[t][d] = rk;
      rv += vc[t][d]; vc[t][d] = rv;
    }
  }
  __syncthreads();
  {  // tail logits: 2 lanes per token
    int t = lane >> 1, half = lane & 1;
    float s = 0.f;
#pragma unroll
    for (int j = 0; j < 32; ++j) {
      int d = half * 32 + j;
      s += bf2f(qb[t * 64 + d]) * kc[t][d];
    }
    s += __shfl_xor(s, 1, 32);
    if (half == 0) tailL[t] = s * scale;
  }
  __syncthreads();
  // logits = q @ Kseg^T : 16 N-tiles of 16 segments, K=64
  v8f L[16];
  const float* Kbh = Kseg + (long)bh * SPAD * 64;
#pragma unroll
  for (int nt = 0; nt < 16; ++nt) {
    v8f acc = {};
#pragma unroll
    for (int kk = 0; kk < 2; ++kk) {
      v16bf a = fragA_lds(qb, 64, kk);
      v16bf bfr = fragB_gf32(Kbh + nt * 16 * 64, 64, kk * 32);
      acc = wmma_bf16(a, bfr, acc);
    }
    L[nt] = acc;
  }
  int col = lane & 15, rbase = (lane >> 4) << 3;
  // scale + causal dyadic mask (t >= r_idx[s])
#pragma unroll
  for (int nt = 0; nt < 16; ++nt) {
    int s = nt * 16 + col;
    int ridx;
    if (s >= SSEG) ridx = 0x7FFFFFFF;
    else {
      int ss = s, cnt = 128, Lsz = 16;
      while (ss >= cnt) { ss -= cnt; cnt >>= 1; Lsz <<= 1; }
      ridx = (ss + 1) * Lsz - 1;
    }
#pragma unroll
    for (int r = 0; r < 8; ++r) {
      int tg = t0 + rbase + r;
      float val = L[nt][r] * scale;
      L[nt][r] = (tg >= ridx) ? val : -INFTY;
    }
  }
  // softmax over 255 segs + tail (row = rbase+r, spread over 16 lanes of half)
#pragma unroll
  for (int r = 0; r < 8; ++r) {
    int row = rbase + r;
    float m = -INFTY;
#pragma unroll
    for (int nt = 0; nt < 16; ++nt) m = fmaxf(m, L[nt][r]);
#pragma unroll
    for (int o = 1; o < 16; o <<= 1) m = fmaxf(m, __shfl_xor(m, o, 32));
    float tl = tailL[row];
    m = fmaxf(m, tl);
    float ssum = 0.f;
#pragma unroll
    for (int nt = 0; nt < 16; ++nt) ssum += __expf(L[nt][r] - m);
#pragma unroll
    for (int o = 1; o < 16; o <<= 1) ssum += __shfl_xor(ssum, o, 32);
    float et = __expf(tl - m);
    ssum += et;
    float inv = 1.f / ssum;
    if (col == 0) tailW[row] = et * inv;
#pragma unroll
    for (int nt = 0; nt < 16; ++nt)
      attw[row * SPAD + nt * 16 + col] = bf16rn(__expf(L[nt][r] - m) * inv);
  }
  __syncthreads();
  // out = att @ Vseg + att_tail * V_tail
  const float* Vbh = VsegT + (long)bh * 64 * SPAD;
  float* outp = aout + (long)(b * TSEQ + t0) * CDIM + h * HS;
#pragma unroll
  for (int e = 0; e < 4; ++e) {
    v8f acc = {};
#pragma unroll
    for (int kk = 0; kk < 8; ++kk) {
      v16bf a = fragA_lds(attw, SPAD, kk);
      v16bf bfr = fragB_gf32(Vbh + (e * 16) * SPAD, SPAD, kk * 32);
      acc = wmma_bf16(a, bfr, acc);
    }
#pragma unroll
    for (int r = 0; r < 8; ++r) {
      int row = rbase + r;
      int eg = e * 16 + col;
      outp[(long)row * CDIM + eg] = acc[r] + tailW[row] * vc[row][eg];
    }
  }
}

extern "C" void kernel_launch(void* const* d_in, const int* in_sizes, int n_in,
                              void* d_out, int out_size, void* d_ws, size_t ws_size,
                              hipStream_t stream) {
  (void)in_sizes; (void)n_in; (void)out_size; (void)ws_size;
  const float* x     = (const float*)d_in[0];
  const float* Wqkv  = (const float*)d_in[1];
  const float* Wproj = (const float*)d_in[2];
  const float* Wv    = (const float*)d_in[3];
  const float* bv    = (const float*)d_in[4];
  float* out = (float*)d_out;

  char* ws = (char*)d_ws;
  size_t off = 0;
  auto alloc = [&](size_t bytes) -> void* {
    void* p = ws + off;
    off = (off + bytes + 255) & ~(size_t)255;
    return p;
  };
  float* qkv   = (float*)alloc((size_t)BB * TSEQ * 3 * CDIM * 4);   // 151 MB
  float* vbuf  = (float*)alloc((size_t)BB * NHEAD * TSEQ * HS * 4); //  50 MB
  float* Kseg  = (float*)alloc((size_t)BB * NHEAD * SPAD * 64 * 4); // 6.3 MB
  float* VsegT = (float*)alloc((size_t)BB * NHEAD * 64 * SPAD * 4); // 6.3 MB
  float* aout  = (float*)alloc((size_t)BB * TSEQ * CDIM * 4);       //  50 MB
  u16*   Wvb   = (u16*)alloc((size_t)HS * HS * HS * 2);             // 0.5 MB

  hipLaunchKernelGGL(k_f32_to_bf16, dim3(1024), dim3(256), 0, stream,
                     Wv, Wvb, HS * HS * HS);
  hipLaunchKernelGGL(k_gemm_bt, dim3(BB * TSEQ / 16, 3 * CDIM / 128), dim3(256),
                     0, stream, x, Wqkv, qkv, CDIM, CDIM, CDIM, 3 * CDIM);
  hipLaunchKernelGGL(k_vproj, dim3(BB * NHEAD * TSEQ / 16), dim3(128), 0, stream,
                     qkv, Wvb, bv, vbuf);
  hipLaunchKernelGGL(k_segsum, dim3(BB * NHEAD), dim3(256), 0, stream,
                     qkv, vbuf, Kseg, VsegT);
  hipLaunchKernelGGL(k_attn, dim3(BB * NHEAD, TSEQ / 16), dim3(32), 0, stream,
                     qkv, vbuf, Kseg, VsegT, aout);
  hipLaunchKernelGGL(k_gemm_bt, dim3(BB * TSEQ / 16, CDIM / 128), dim3(256),
                     0, stream, aout, Wproj, out, CDIM, CDIM, CDIM, CDIM);
}
